// VQVAETokenizer_498216206955
// MI455X (gfx1250) — matrix operand order repaired and checked by reading, not compiled, not run on hardware
//
#include <hip/hip_runtime.h>

// ---------------------------------------------------------------------------
// VQ-VAE tokenizer forward for gfx1250 (MI455X).
// conv1/conv2/conv3 as implicit-GEMM with v_wmma_f32_16x16x32_bf16.
// - weights pre-packed into per-32K-block fragment-native layout, B fragments
//   loaded straight from global (L2-resident) as 2x b128 per lane
// - A staged through double-buffered LDS (vectorized im2col gather, packed
//   v_cvt_pk_bf16_f32 conversion), fragments as 2x ds_load_b128
// - wave tile 32Mx64N -> 8 WMMA per 32-K step per wave
// Scalar VQ argmin over 512 codes, z_q == codebook[idx],
// loss = 1.25 * mean((z_q - z)^2).
// ---------------------------------------------------------------------------

typedef __attribute__((ext_vector_type(16))) __bf16 v16bf;
typedef __attribute__((ext_vector_type(8)))  float  v8f;

union AFrag { v16bf v; __bf16 s[16]; uint4 q[2]; };
union CFrag { v8f   v; float  s[8];  };

// packed f32x2 -> bf16x2 (single VALU op, round-to-nearest-even)
__device__ __forceinline__ unsigned pk_bf16(float a, float b) {
    unsigned r;
    asm volatile("v_cvt_pk_bf16_f32 %0, %1, %2" : "=v"(r) : "v"(a), "v"(b));
    return r;
}

// ---------------------------------------------------------------------------
// Weight packing: W[O][C][kd][kh][kw] (f32) -> blocked bf16
//   Bwp[(kb*NOUT + o)*32 + kl],  k = kb*32 + kl = ((kd*4+kh)*4+kw)*C + c
// so each lane's WMMA B fragment (16 consecutive k) is 32 contiguous bytes.
// ---------------------------------------------------------------------------
__global__ void pack_w1(const float* __restrict__ W1, __bf16* __restrict__ Bwp) {
    int t = blockIdx.x * 256 + threadIdx.x;        // t = k*128 + o
    if (t >= 512 * 128) return;
    int k = t >> 7, o = t & 127;
    int q = k >> 3, c = k & 7;
    int dz = q >> 4, dy = (q >> 2) & 3, dx = q & 3;
    Bwp[(((size_t)(k >> 5) * 128 + o) << 5) + (k & 31)] =
        (__bf16)W1[(((o * 8 + c) * 4 + dz) * 4 + dy) * 4 + dx];
}

__global__ void pack_w2(const float* __restrict__ W2, __bf16* __restrict__ Bwp) {
    int t = blockIdx.x * 256 + threadIdx.x;        // t = k*128 + o, k < 8192
    if (t >= 8192 * 128) return;
    int k = t >> 7, o = t & 127;
    int q = k >> 7, c = k & 127;
    int dz = q >> 4, dy = (q >> 2) & 3, dx = q & 3;
    Bwp[(((size_t)(k >> 5) * 128 + o) << 5) + (k & 31)] =
        (__bf16)W2[(((o * 128 + c) * 4 + dz) * 4 + dy) * 4 + dx];
}

__global__ void pack_w3(const float* __restrict__ W3, __bf16* __restrict__ Bwp) {
    int t = blockIdx.x * 256 + threadIdx.x;        // t = k*64 + o, k < 128
    if (t >= 128 * 64) return;
    int k = t >> 6, o = t & 63;
    Bwp[(((size_t)(k >> 5) * 64 + o) << 5) + (k & 31)] = (__bf16)W3[o * 128 + k];
}

// codebook norms + zero the loss accumulator
__global__ void vq_prep(const float* __restrict__ cb, float* __restrict__ cn2,
                        float* __restrict__ loss_acc) {
    int t = blockIdx.x * 256 + threadIdx.x;
    if (t == 0) loss_acc[0] = 0.f;
    if (t < 512) {
        float s = 0.f;
        #pragma unroll 8
        for (int e = 0; e < 64; ++e) { float v = cb[t * 64 + e]; s += v * v; }
        cn2[t] = s;
    }
}

// ---------------------------------------------------------------------------
// A-tile staging: each thread stages 16 k-elements of one M-row into LDS.
// One bounds check per conv tap, b128 global loads, b128 LDS stores.
// ---------------------------------------------------------------------------
template <int LAYER>
__device__ __forceinline__ void stage_A(
    __bf16* __restrict__ As,                     // [128][ASTR=40]
    const float* __restrict__ in_f32, const __bf16* __restrict__ in_bf16,
    int k0, int a_row, int a_koff, int vn, int od, int oh, int ow, int am)
{
    constexpr int ASTR = 40;
    uint4 t0 = {0, 0, 0, 0}, t1 = {0, 0, 0, 0};
    if constexpr (LAYER == 1) {
        // 16 k = 2 taps x 8 input channels (channels-last, contiguous f32)
        uint4* outq[2] = {&t0, &t1};
        #pragma unroll
        for (int p = 0; p < 2; ++p) {
            const int q  = ((k0 + a_koff) >> 3) + p;
            const int dz = q >> 4, dy = (q >> 2) & 3, dx = q & 3;
            const int id = od * 2 - 1 + dz, ih = oh * 2 - 1 + dy, iw = ow * 2 - 1 + dx;
            if ((unsigned)id < 16u && (unsigned)ih < 128u && (unsigned)iw < 128u) {
                const float4* src =
                    (const float4*)&in_f32[(((size_t)(vn * 16 + id) * 128 + ih) * 128 + iw) * 8];
                const float4 f0 = src[0], f1 = src[1];
                outq[p]->x = pk_bf16(f0.x, f0.y);   // packed cvt: 4 ops / 8 elems
                outq[p]->y = pk_bf16(f0.z, f0.w);
                outq[p]->z = pk_bf16(f1.x, f1.y);
                outq[p]->w = pk_bf16(f1.z, f1.w);
            }
        }
    } else if constexpr (LAYER == 2) {
        // 16 k = 16 channels of a single tap (bf16, contiguous, 32B aligned)
        const int kb = k0 + a_koff;
        const int q  = kb >> 7, c0 = kb & 127;
        const int dz = q >> 4, dy = (q >> 2) & 3, dx = q & 3;
        const int id = od * 2 - 1 + dz, ih = oh * 2 - 1 + dy, iw = ow * 2 - 1 + dx;
        if ((unsigned)id < 8u && (unsigned)ih < 64u && (unsigned)iw < 64u) {
            const uint4* src =
                (const uint4*)&in_bf16[(((size_t)(vn * 8 + id) * 64 + ih) * 64 + iw) * 128 + c0];
            t0 = src[0]; t1 = src[1];
        }
    } else {
        // 1x1x1 conv: plain contiguous row
        const uint4* src = (const uint4*)&in_bf16[(size_t)am * 128 + k0 + a_koff];
        t0 = src[0]; t1 = src[1];
    }
    uint4* dst = (uint4*)&As[a_row * ASTR + a_koff];    // 16B aligned (80B row)
    dst[0] = t0;
    dst[1] = t1;
}

// ---------------------------------------------------------------------------
// Implicit-GEMM conv. 256 threads = 8 waves (4x2), block tile 128(M) x BN x 32(K),
// wave tile 32(M) x 16*JN(N): JN=4 (BN=128, layers 1-2), JN=2 (BN=64, layer 3).
//   LAYER 1: x (8,16,128,128,8) f32   -> h1 (8,8,64,64,128) bf16, ReLU, K=512
//   LAYER 2: h1 (8,8,64,64,128) bf16  -> h2 (8,4,32,32,128) bf16, ReLU, K=8192
//   LAYER 3: h2 flat (32768,128) bf16 -> z  (32768,64)      f32,        K=128
// ---------------------------------------------------------------------------
template <int LAYER>
__global__ __launch_bounds__(256) void conv_gemm_kernel(
    const float*  __restrict__ in_f32,
    const __bf16* __restrict__ in_bf16,
    const __bf16* __restrict__ Bwp,    // blocked [K/32][NOUT][32] bf16
    const float*  __restrict__ bias,   // [NOUT]
    __bf16*       __restrict__ out_bf16,
    float*        __restrict__ out_f32)
{
    constexpr int BM = 128, BK = 32;
    constexpr int K    = (LAYER == 1) ? 512 : (LAYER == 2) ? 8192 : 128;
    constexpr int NOUT = (LAYER == 3) ? 64 : 128;
    constexpr int JN   = (LAYER == 3) ? 2 : 4;   // 16-col subtiles per wave
    constexpr int ASTR = 40;   // 80B row stride: 16B aligned, 20-dword bank step

    __shared__ __bf16 As[2][BM * ASTR];   // double-buffered A tile (20 KB)

    const int tid  = threadIdx.x;
    const int lane = tid & 31;
    const int wave = tid >> 5;
    const int wm   = wave >> 1;          // 0..3  (32-row subtile)
    const int wn   = wave & 1;           // 0..1  (JN*16-col subtile)
    const int m0   = blockIdx.x * BM;

    // A loader: 2 threads per row, 16 k-elements each
    const int a_row  = tid >> 1;
    const int a_koff = (tid & 1) * 16;
    const int am     = m0 + a_row;

    // loop-invariant voxel decode for the im2col gather
    int vn = 0, od = 0, oh = 0, ow = 0;
    if constexpr (LAYER == 1) {
        vn = am >> 15; int r = am & 32767;
        od = r >> 12; oh = (r >> 6) & 63; ow = r & 63;
    } else if constexpr (LAYER == 2) {
        vn = am >> 12; int r = am & 4095;
        od = r >> 10; oh = (r >> 5) & 31; ow = r & 31;
    }

    const int rl = lane & 15;
    const int hi = lane >> 4;
    const int oc0 = wn * (16 * JN) + rl;          // this lane's first B column

    CFrag acc[2][JN];
    #pragma unroll
    for (int i = 0; i < 2; ++i)
        #pragma unroll
        for (int j = 0; j < JN; ++j)
            #pragma unroll
            for (int r = 0; r < 8; ++r) acc[i][j].s[r] = 0.f;

    stage_A<LAYER>(As[0], in_f32, in_bf16, 0, a_row, a_koff, vn, od, oh, ow, am);
    int buf = 0;

    for (int k0 = 0; k0 < K; k0 += BK) {
        __syncthreads();                              // staged tile visible
        if (k0 + BK < K)                              // overlap: stage next tile
            stage_A<LAYER>(As[buf ^ 1], in_f32, in_bf16, k0 + BK,
                           a_row, a_koff, vn, od, oh, ow, am);

        // ---- B fragments straight from global (fragment-native layout) ----
        // lane col o, elements e -> klocal = e + hi*16: 32 contiguous bytes
        AFrag b[JN];
        #pragma unroll
        for (int j = 0; j < JN; ++j) {
            const uint4* bp = (const uint4*)
                &Bwp[(((size_t)(k0 >> 5) * NOUT + oc0 + j * 16) << 5) + hi * 16];
            b[j].q[0] = bp[0];
            b[j].q[1] = bp[1];
        }
        if (k0 + 2 * BK < K)   // gfx1250 global_prefetch_b8 on the weight stream
            __builtin_prefetch(
                &Bwp[(((size_t)((k0 >> 5) + 2) * NOUT + oc0) << 5)], 0, 0);

        // ---- A fragments: 2x ds_load_b128 per lane (contiguous k = hi*16 + e, same as B) ----
        AFrag a[2];
        #pragma unroll
        for (int i = 0; i < 2; ++i) {
            const __bf16* base = &As[buf][(wm * 32 + i * 16 + rl) * ASTR];
            a[i].q[0] = *(const uint4*)&base[hi * 16];       // e 0..7  -> K hi*16..
            a[i].q[1] = *(const uint4*)&base[hi * 16 + 8];   // e 8..15 -> K hi*16+8..
        }

        #pragma unroll
        for (int i = 0; i < 2; ++i)
            #pragma unroll
            for (int j = 0; j < JN; ++j)
                acc[i][j].v = __builtin_amdgcn_wmma_f32_16x16x32_bf16(
                    false, a[i].v, false, b[j].v, (short)0, acc[i][j].v, false, false);

        buf ^= 1;
    }

    // -------- epilogue: bias (+ReLU), store channels-last --------
    #pragma unroll
    for (int i = 0; i < 2; ++i) {
        const int rowb = m0 + wm * 32 + i * 16 + hi * 8;   // C: VGPR r -> row r (+8 hi)
        #pragma unroll
        for (int j = 0; j < JN; ++j) {
            const int col = oc0 + j * 16;
            const float bv = bias[col];
            #pragma unroll
            for (int r = 0; r < 8; ++r) {
                const int row = rowb + r;
                float vx = acc[i][j].s[r] + bv;
                if constexpr (LAYER != 3) {
                    vx = vx > 0.f ? vx : 0.f;
                    out_bf16[(size_t)row * NOUT + col] = (__bf16)vx;
                } else {
                    out_f32[(size_t)row * NOUT + col] = vx;
                }
            }
        }
    }
}

// ---------------------------------------------------------------------------
// VQ: per-voxel argmin over 512 codes (row-constant |z|^2 dropped; strict <
// keeps jnp.argmin first-minimum semantics), gather z_q, accumulate loss.
// ---------------------------------------------------------------------------
__global__ __launch_bounds__(256) void vq_kernel(
    const float* __restrict__ z, const float* __restrict__ cb,
    const float* __restrict__ cn2, float* __restrict__ zq,
    int* __restrict__ idx, float* __restrict__ loss_acc)
{
    const int m = blockIdx.x * 256 + threadIdx.x;   // grid sized exactly: m < 32768
    float zr[64];
    #pragma unroll
    for (int e = 0; e < 64; e += 4) {
        float4 t = *(const float4*)(z + (size_t)m * 64 + e);
        zr[e] = t.x; zr[e + 1] = t.y; zr[e + 2] = t.z; zr[e + 3] = t.w;
    }
    float best = 3.4e38f; int bi = 0;
    for (int kc = 0; kc < 512; ++kc) {
        const float4* c4 = (const float4*)(cb + (size_t)kc * 64);
        float dot = 0.f;
        #pragma unroll
        for (int e = 0; e < 16; ++e) {
            float4 c = c4[e];
            dot += zr[4 * e] * c.x + zr[4 * e + 1] * c.y
                 + zr[4 * e + 2] * c.z + zr[4 * e + 3] * c.w;
        }
        const float d = cn2[kc] - 2.f * dot;
        if (d < best) { best = d; bi = kc; }
    }
    idx[m] = bi;
    float ls = 0.f;
    #pragma unroll
    for (int e = 0; e < 64; ++e) {
        const float q = cb[(size_t)bi * 64 + e];
        zq[(size_t)m * 64 + e] = q;                 // straight-through output
        const float df = q - zr[e];
        ls += df * df;
    }
    for (int off = 16; off > 0; off >>= 1) ls += __shfl_down(ls, off, 32);
    if ((threadIdx.x & 31) == 0) atomicAdd(loss_acc, ls);
}

__global__ void loss_finalize(const float* __restrict__ acc, float* __restrict__ out) {
    // mean((sg(zq)-z)^2) + 0.25*mean((zq-sg(z))^2) == 1.25 * mean((zq-z)^2)
    out[0] = 1.25f * acc[0] / (float)(32768 * 64);
}

// ---------------------------------------------------------------------------
extern "C" void kernel_launch(void* const* d_in, const int* in_sizes, int n_in,
                              void* d_out, int out_size, void* d_ws, size_t ws_size,
                              hipStream_t stream)
{
    const float* x  = (const float*)d_in[0];   // (8,16,128,128,8)
    const float* W1 = (const float*)d_in[1];   // (128,8,4,4,4)
    const float* b1 = (const float*)d_in[2];
    const float* W2 = (const float*)d_in[3];   // (128,128,4,4,4)
    const float* b2 = (const float*)d_in[4];
    const float* W3 = (const float*)d_in[5];   // (64,128,1,1,1)
    const float* b3 = (const float*)d_in[6];
    const float* cb = (const float*)d_in[7];   // (512,64)

    constexpr size_t M1 = 262144;  // 8*8*64*64
    constexpr size_t M2 = 32768;   // 8*4*32*32
    constexpr size_t E  = 64;

    float* zq_out   = (float*)d_out;                 // (32768, 64) f32
    float* loss_out = zq_out + M2 * E;               // scalar
    int*   idx_out  = (int*)(zq_out + M2 * E + 1);   // (32768,) int32

    // workspace carve (256B aligned), ~86 MB total
    char* p = (char*)d_ws;
    auto carve = [&](size_t bytes) -> char* {
        char* r = p; p += (bytes + 255) & ~(size_t)255; return r;
    };
    float*  loss_acc = (float*)carve(sizeof(float));
    float*  cn2      = (float*)carve(512 * sizeof(float));
    __bf16* Bw1      = (__bf16*)carve((size_t)512 * 128 * sizeof(__bf16));
    __bf16* Bw2      = (__bf16*)carve((size_t)8192 * 128 * sizeof(__bf16));
    __bf16* Bw3      = (__bf16*)carve((size_t)128 * 64 * sizeof(__bf16));
    __bf16* h1       = (__bf16*)carve(M1 * 128 * sizeof(__bf16));
    __bf16* h2       = (__bf16*)carve(M2 * 128 * sizeof(__bf16));
    float*  zf       = (float*)carve(M2 * E * sizeof(float));

    pack_w1<<<(512 * 128 + 255) / 256, 256, 0, stream>>>(W1, Bw1);
    pack_w2<<<(8192 * 128 + 255) / 256, 256, 0, stream>>>(W2, Bw2);
    pack_w3<<<(128 * 64 + 255) / 256, 256, 0, stream>>>(W3, Bw3);
    vq_prep<<<2, 256, 0, stream>>>(cb, cn2, loss_acc);

    conv_gemm_kernel<1><<<dim3(2048, 1), 256, 0, stream>>>(x, nullptr, Bw1, b1, h1, nullptr);
    conv_gemm_kernel<2><<<dim3(256, 1),  256, 0, stream>>>(nullptr, h1, Bw2, b2, h2, nullptr);
    conv_gemm_kernel<3><<<dim3(256, 1),  256, 0, stream>>>(nullptr, h2, Bw3, b3, nullptr, zf);

    vq_kernel<<<128, 256, 0, stream>>>(zf, cb, cn2, zq_out, idx_out, loss_acc);
    loss_finalize<<<1, 1, 0, stream>>>(loss_acc, loss_out);
}
